// ClientEncoder_47734266527805
// MI455X (gfx1250) — matrix operand order repaired and checked by reading, not compiled, not run on hardware
//
#include <hip/hip_runtime.h>
#include <math.h>

#define DEV __device__ __forceinline__

typedef __attribute__((ext_vector_type(16))) __bf16 v16bf_t;
typedef __attribute__((ext_vector_type(8)))  float  v8f_t;
typedef __attribute__((ext_vector_type(4)))  unsigned int v4u_t;

DEV unsigned short f2bf(float f) {
  unsigned int u = __float_as_uint(f);
  u += 0x7FFFu + ((u >> 16) & 1u);   // round-to-nearest-even
  return (unsigned short)(u >> 16);
}

DEV v8f_t v8f_zero() {
  v8f_t z;
#pragma unroll
  for (int i = 0; i < 8; ++i) z[i] = 0.f;
  return z;
}

// pack 8 f32 -> 8 bf16 and store as one 16B LDS write
DEV void store8_bf16(unsigned short* dst, const float* v) {
  union { unsigned short u[8]; v4u_t q; } t;
#pragma unroll
  for (int i = 0; i < 8; ++i) t.u[i] = f2bf(v[i]);
  *(v4u_t*)dst = t.q;
}

// A fragment (16x32 bf16) from LDS tile stored [row][k], stride in ushorts (16B aligned).
// lanes 0-15: M=lane, K=0..15; lanes 16-31: M=lane-16, K=16..31
// -> per lane: one 32B contiguous run => 2x ds_load_b128
DEV v16bf_t load_afrag(const unsigned short* S, int stride, int row0, int koff, int lane) {
  const unsigned short* p =
      S + (row0 + (lane & 15)) * stride + koff + ((lane < 16) ? 0 : 16);
  union { v4u_t q[2]; v16bf_t v; } t;
  t.q[0] = *(const v4u_t*)p;
  t.q[1] = *(const v4u_t*)(p + 8);
  return t.v;
}

// B fragment (32x16 bf16) from LDS tile stored [n][k], stride in ushorts.
// lanes 0-15: N=lane, K=0..15; lanes 16-31: N=lane-16, K=16..31
// -> per lane: one 32B contiguous run => 2x ds_load_b128
DEV v16bf_t load_bfrag(const unsigned short* S, int stride, int n0, int lane) {
  const unsigned short* p =
      S + (n0 + (lane & 15)) * stride + ((lane < 16) ? 0 : 16);
  union { v4u_t q[2]; v16bf_t v; } t;
  t.q[0] = *(const v4u_t*)p;
  t.q[1] = *(const v4u_t*)(p + 8);
  return t.v;
}

DEV v8f_t wmma_bf16(v16bf_t a, v16bf_t b, v8f_t c) {
  return __builtin_amdgcn_wmma_f32_16x16x32_bf16(false, a, false, b, (short)0, c, false, false);
}

// fetch 16 consecutive K-values of one logical row (guarded, zero-filled)
DEV void fetch_row16(float* r, const float* P, int ld, int nrows, int K,
                     int grow, int kbase) {
  const float* p = P + (size_t)grow * ld + kbase;
#pragma unroll
  for (int i = 0; i < 16; ++i) {
    int gk = kbase + i;
    r[i] = (grow < nrows && gk < K) ? p[i] : 0.f;
  }
}

// ---------------------------------------------------------------------------
// Generic GEMM: C[M,N] = A[M,K](f32->bf16) * op(B)(f32->bf16) (+bias) (+C if ACCUM)
// BTRANS: B stored [N,K] (A @ B^T). BIAS_MODE: 0 none, 1 per-col(n), 2 per-row(m).
// 128x128 tile, 8 wave32 waves, K-step 32, register-prefetch pipeline.
// ---------------------------------------------------------------------------
template <int BIAS_MODE, bool BTRANS, bool ACCUM>
__global__ __launch_bounds__(256) void gemm_bf16_kernel(
    const float* __restrict__ A, const float* __restrict__ B,
    const float* __restrict__ bias, float* __restrict__ C,
    int M, int N, int K, int lda, int ldb, int ldc)
{
  __shared__ __align__(16) unsigned short As[128 * 40];  // [m][k], 80B rows
  __shared__ __align__(16) unsigned short Bs[128 * 40];  // [n][k]

  const int tid  = threadIdx.x;
  const int lane = tid & 31;
  const int wave = tid >> 5;
  const int m0 = blockIdx.y * 128;
  const int n0 = blockIdx.x * 128;
  const int wm = (wave & 3) * 32;   // 4 waves along M
  const int wn = (wave >> 2) * 64;  // 2 waves along N

  // staging thread mapping
  const int a_row = tid >> 1;             // 0..127
  const int a_ks  = (tid & 1) * 16;       // 0 / 16
  const int gmA   = m0 + a_row;
  // BTRANS path: same shape over B rows (n)
  const int gnB   = n0 + a_row;
  // !BTRANS path: read along n, write transposed
  const int b_krow = tid >> 3;            // 0..31
  const int b_ns   = (tid & 7) * 16;      // 0..112

  v8f_t acc[2][4];
#pragma unroll
  for (int i = 0; i < 2; ++i)
#pragma unroll
    for (int j = 0; j < 4; ++j) acc[i][j] = v8f_zero();

  float a_reg[16], b_reg[16];
  fetch_row16(a_reg, A, lda, M, K, gmA, a_ks);
  if (BTRANS) {
    fetch_row16(b_reg, B, ldb, N, K, gnB, a_ks);
  } else {
    const float* bp = B + (size_t)b_krow * ldb + (n0 + b_ns);
#pragma unroll
    for (int i = 0; i < 16; ++i) {
      int gn = n0 + b_ns + i;
      b_reg[i] = (b_krow < K && gn < N) ? bp[i] : 0.f;
    }
  }

  for (int k0 = 0; k0 < K; k0 += 32) {
    // ---- stage current K-slab into LDS ----
    store8_bf16(&As[a_row * 40 + a_ks], a_reg);
    store8_bf16(&As[a_row * 40 + a_ks + 8], a_reg + 8);
    if (BTRANS) {
      store8_bf16(&Bs[a_row * 40 + a_ks], b_reg);
      store8_bf16(&Bs[a_row * 40 + a_ks + 8], b_reg + 8);
    } else {
#pragma unroll
      for (int i = 0; i < 16; ++i) Bs[(b_ns + i) * 40 + b_krow] = f2bf(b_reg[i]);
    }
    __syncthreads();

    // ---- prefetch next K-slab into registers (overlaps with WMMA below) ----
    const int k1 = k0 + 32;
    if (k1 < K) {
      fetch_row16(a_reg, A, lda, M, K, gmA, k1 + a_ks);
      if (BTRANS) {
        fetch_row16(b_reg, B, ldb, N, K, gnB, k1 + a_ks);
      } else {
        int gk = k1 + b_krow;
        const float* bp = B + (size_t)gk * ldb + (n0 + b_ns);
#pragma unroll
        for (int i = 0; i < 16; ++i) {
          int gn = n0 + b_ns + i;
          b_reg[i] = (gk < K && gn < N) ? bp[i] : 0.f;
        }
      }
    }

    // ---- compute ----
    v16bf_t bfr[4];
#pragma unroll
    for (int j = 0; j < 4; ++j) bfr[j] = load_bfrag(Bs, 40, wn + 16 * j, lane);
#pragma unroll
    for (int i = 0; i < 2; ++i) {
      v16bf_t af = load_afrag(As, 40, wm + 16 * i, 0, lane);
#pragma unroll
      for (int j = 0; j < 4; ++j) acc[i][j] = wmma_bf16(af, bfr[j], acc[i][j]);
    }
    __syncthreads();
  }

  // epilogue: C layout lane -> N=lane&15, VGPR v -> M = v (+8 for hi lanes)
#pragma unroll
  for (int i = 0; i < 2; ++i) {
#pragma unroll
    for (int j = 0; j < 4; ++j) {
      int gn = n0 + wn + 16 * j + (lane & 15);
      int mb = m0 + wm + 16 * i + ((lane < 16) ? 0 : 8);
      if (gn < N) {
#pragma unroll
        for (int v = 0; v < 8; ++v) {
          int gm = mb + v;
          if (gm < M) {
            float r = acc[i][j][v];
            if (BIAS_MODE == 1) r += bias[gn];
            if (BIAS_MODE == 2) r += bias[gm];
            size_t off = (size_t)gm * ldc + gn;
            if (ACCUM) r += C[off];
            C[off] = r;
          }
        }
      }
    }
  }
}

// ---------------------------------------------------------------------------
// Per-(b,f) time-series normalization; writes x transposed to [bf][t]
// ---------------------------------------------------------------------------
__global__ __launch_bounds__(256) void norm_kernel(const float* __restrict__ x,
                                                   const float* __restrict__ mask,
                                                   float* __restrict__ xn)
{
  __shared__ float sA[256];
  __shared__ float sB[256];
  __shared__ float s_mean, s_inv;
  const int bf = blockIdx.x;
  const int b = bf / 7, f = bf % 7;
  const int tid = threadIdx.x;

  float sx = 0.f, sc = 0.f;
  for (int t = tid; t < 512; t += 256) {
    size_t o = ((size_t)b * 512 + t) * 7 + f;
    sx += x[o];
    sc += mask[o];  // mask is 0/1
  }
  sA[tid] = sx; sB[tid] = sc;
  __syncthreads();
  for (int off = 128; off > 0; off >>= 1) {
    if (tid < off) { sA[tid] += sA[tid + off]; sB[tid] += sB[tid + off]; }
    __syncthreads();
  }
  if (tid == 0) s_mean = sA[0] / sB[0];
  __syncthreads();
  const float mean = s_mean;
  const float cnt  = sB[0];
  __syncthreads();

  float ss = 0.f;
  for (int t = tid; t < 512; t += 256) {
    size_t o = ((size_t)b * 512 + t) * 7 + f;
    float mv = mask[o];
    float xv = (mv != 0.f) ? (x[o] - mean) : 0.f;
    ss += xv * xv;
  }
  sA[tid] = ss;
  __syncthreads();
  for (int off = 128; off > 0; off >>= 1) {
    if (tid < off) sA[tid] += sA[tid + off];
    __syncthreads();
  }
  if (tid == 0) s_inv = 1.f / sqrtf(sA[0] / cnt + 1e-5f);
  __syncthreads();
  const float inv = s_inv;
  for (int t = tid; t < 512; t += 256) {
    size_t o = ((size_t)b * 512 + t) * 7 + f;
    float mv = mask[o];
    xn[(size_t)bf * 512 + t] = (mv != 0.f) ? (x[o] - mean) * inv : 0.f;
  }
}

// ---------------------------------------------------------------------------
// Patch embedding (K=16 -> scalar FMA): x_emb / m_emb for one token per block
// ---------------------------------------------------------------------------
__global__ __launch_bounds__(256) void patch_embed_kernel(
    const float* __restrict__ xn, const float* __restrict__ mask,
    const float* __restrict__ Wf, const float* __restrict__ bfeat,
    const float* __restrict__ Wb, const float* __restrict__ bbin,
    float* __restrict__ xe, float* __restrict__ me)
{
  const int p  = blockIdx.x;   // 0..62
  const int bf = blockIdx.y;   // 0..111
  const int b = bf / 7, f = bf % 7;
  __shared__ float xp[16];
  __shared__ float mp[16];
  if (threadIdx.x < 16) {
    int t = p * 8 + threadIdx.x;
    xp[threadIdx.x] = xn[(size_t)bf * 512 + t];
    mp[threadIdx.x] = mask[((size_t)b * 512 + t) * 7 + f];
  }
  __syncthreads();
  size_t row = (size_t)(bf * 63 + p) * 768;
  for (int d = threadIdx.x; d < 768; d += 256) {
    float a1 = bfeat[d], a2 = bbin[d];
    const float* w1 = Wf + d * 16;
    const float* w2 = Wb + d * 16;
#pragma unroll
    for (int hh = 0; hh < 16; ++hh) {
      a1 += xp[hh] * w1[hh];
      a2 += mp[hh] * w2[hh];
    }
    xe[row + d] = a1;
    me[row + d] = a2;
  }
}

// h = sigmoid(g)*xe + (1-sigmoid(g))*me, in place over g
__global__ void gate_kernel(float* __restrict__ g, const float* __restrict__ xe,
                            const float* __restrict__ me, size_t n)
{
  for (size_t i = (size_t)blockIdx.x * blockDim.x + threadIdx.x; i < n;
       i += (size_t)gridDim.x * blockDim.x) {
    float s = 1.f / (1.f + __expf(-g[i]));
    g[i] = s * xe[i] + (1.f - s) * me[i];
  }
}

// ---------------------------------------------------------------------------
// Fused attention statistics per (bf, h): WMMA scores + 3-pass softmax-column-sum
// tmps[bf,h,s] = sum_l softmax_s(q_l . k_s / 8)
// ---------------------------------------------------------------------------
// B fragment for k: from global f32 (L2 resident), lane N = token s, K = head dim e
DEV v16bf_t load_kfrag(const float* kh, int s0, int ke, int lane) {
  union { unsigned short u[16]; v16bf_t v; } t;
  int s  = s0 + (lane & 15);
  int eb = ke + ((lane < 16) ? 0 : 16);
  if (s < 1000) {
    const float4* p = (const float4*)(kh + (size_t)s * 768 + eb);
#pragma unroll
    for (int w = 0; w < 4; ++w) {
      float4 f = p[w];
      t.u[4 * w + 0] = f2bf(f.x);
      t.u[4 * w + 1] = f2bf(f.y);
      t.u[4 * w + 2] = f2bf(f.z);
      t.u[4 * w + 3] = f2bf(f.w);
    }
  } else {
#pragma unroll
    for (int v = 0; v < 16; ++v) t.u[v] = 0;
  }
  return t.v;
}

__global__ __launch_bounds__(128) void attn_scores_kernel(
    const float* __restrict__ qbuf, const float* __restrict__ kbuf,
    float* __restrict__ tmps)
{
  __shared__ __align__(16) unsigned short qs[64 * 72];  // q rows bf16, 144B stride
  __shared__ float red[64 * 64];                        // per-row reduction scratch
  __shared__ float rowmax[64];
  __shared__ float rowsum[64];
  __shared__ float tl[1008];

  const int bf = blockIdx.x;   // 0..111
  const int h  = blockIdx.y;   // 0..11
  const int tid  = threadIdx.x;
  const int lane = tid & 31;
  const int wave = tid >> 5;

  // stage q (packed b128 stores)
  for (int g = tid; g < 64 * 8; g += 128) {
    int l = g >> 3, e0 = (g & 7) * 8;
    float tmp[8];
    if (l < 63) {
      const float* qp = qbuf + (size_t)(bf * 63 + l) * 768 + h * 64 + e0;
#pragma unroll
      for (int i = 0; i < 8; ++i) tmp[i] = qp[i];
    } else {
#pragma unroll
      for (int i = 0; i < 8; ++i) tmp[i] = 0.f;
    }
    store8_bf16(&qs[l * 72 + e0], tmp);
  }
  for (int i = tid; i < 1008; i += 128) tl[i] = 0.f;
  __syncthreads();

  const float* kh = kbuf + h * 64;
  const int sl   = lane & 15;
  const int rofs = (lane < 16) ? 0 : 8;
  const int cid  = wave * 16 + sl;

  float lst[4][8];

  // ---- pass 1: row max ----
#pragma unroll
  for (int f = 0; f < 4; ++f)
#pragma unroll
    for (int v = 0; v < 8; ++v) lst[f][v] = -3.4e38f;

  for (int ts = wave; ts < 63; ts += 4) {
    int s0 = ts * 16;
    v8f_t c[4];
#pragma unroll
    for (int f = 0; f < 4; ++f) c[f] = v8f_zero();
#pragma unroll
    for (int ks = 0; ks < 2; ++ks) {
      v16bf_t bfr = load_kfrag(kh, s0, ks * 32, lane);
#pragma unroll
      for (int f = 0; f < 4; ++f) {
        v16bf_t af = load_afrag(qs, 72, f * 16, ks * 32, lane);
        c[f] = wmma_bf16(af, bfr, c[f]);
      }
    }
    if (s0 + sl < 1000) {
#pragma unroll
      for (int f = 0; f < 4; ++f)
#pragma unroll
        for (int v = 0; v < 8; ++v) lst[f][v] = fmaxf(lst[f][v], c[f][v] * 0.125f);
    }
  }
#pragma unroll
  for (int f = 0; f < 4; ++f)
#pragma unroll
    for (int v = 0; v < 8; ++v) red[(f * 16 + rofs + v) * 64 + cid] = lst[f][v];
  __syncthreads();
  if (tid < 64) {
    float m = -3.4e38f;
    for (int c2 = 0; c2 < 64; ++c2) m = fmaxf(m, red[tid * 64 + c2]);
    rowmax[tid] = m;
  }
  __syncthreads();

  // ---- pass 2: row sum of exp ----
#pragma unroll
  for (int f = 0; f < 4; ++f)
#pragma unroll
    for (int v = 0; v < 8; ++v) lst[f][v] = 0.f;

  for (int ts = wave; ts < 63; ts += 4) {
    int s0 = ts * 16;
    v8f_t c[4];
#pragma unroll
    for (int f = 0; f < 4; ++f) c[f] = v8f_zero();
#pragma unroll
    for (int ks = 0; ks < 2; ++ks) {
      v16bf_t bfr = load_kfrag(kh, s0, ks * 32, lane);
#pragma unroll
      for (int f = 0; f < 4; ++f) {
        v16bf_t af = load_afrag(qs, 72, f * 16, ks * 32, lane);
        c[f] = wmma_bf16(af, bfr, c[f]);
      }
    }
    if (s0 + sl < 1000) {
#pragma unroll
      for (int f = 0; f < 4; ++f)
#pragma unroll
        for (int v = 0; v < 8; ++v) {
          int l = f * 16 + rofs + v;
          lst[f][v] += __expf(c[f][v] * 0.125f - rowmax[l]);
        }
    }
  }
#pragma unroll
  for (int f = 0; f < 4; ++f)
#pragma unroll
    for (int v = 0; v < 8; ++v) red[(f * 16 + rofs + v) * 64 + cid] = lst[f][v];
  __syncthreads();
  if (tid < 64) {
    float s = 0.f;
    for (int c2 = 0; c2 < 64; ++c2) s += red[tid * 64 + c2];
    rowsum[tid] = s;
  }
  __syncthreads();

  // ---- pass 3: accumulate softmax column sums ----
  for (int ts = wave; ts < 63; ts += 4) {
    int s0 = ts * 16;
    v8f_t c[4];
#pragma unroll
    for (int f = 0; f < 4; ++f) c[f] = v8f_zero();
#pragma unroll
    for (int ks = 0; ks < 2; ++ks) {
      v16bf_t bfr = load_kfrag(kh, s0, ks * 32, lane);
#pragma unroll
      for (int f = 0; f < 4; ++f) {
        v16bf_t af = load_afrag(qs, 72, f * 16, ks * 32, lane);
        c[f] = wmma_bf16(af, bfr, c[f]);
      }
    }
    if (s0 + sl < 1000) {
      float part = 0.f;
#pragma unroll
      for (int f = 0; f < 4; ++f)
#pragma unroll
        for (int v = 0; v < 8; ++v) {
          int l = f * 16 + rofs + v;
          if (l < 63) part += __expf(c[f][v] * 0.125f - rowmax[l]) / rowsum[l];
        }
      atomicAdd(&tl[s0 + sl], part);
    }
  }
  __syncthreads();
  for (int s = tid; s < 1000; s += 128)
    tmps[((size_t)bf * 12 + h) * 1000 + s] = tl[s];
}

// ---------------------------------------------------------------------------
// top-8 per (bf,h) by iterative argmax (ties -> lowest index)
// ---------------------------------------------------------------------------
__global__ __launch_bounds__(128) void topk_kernel(const float* __restrict__ tmps,
                                                   int* __restrict__ idxs)
{
  __shared__ float vals[1000];
  __shared__ float rv[128];
  __shared__ int   ri[128];
  const int bh = blockIdx.x;  // 0..1343
  const int tid = threadIdx.x;
  for (int i = tid; i < 1000; i += 128) vals[i] = tmps[(size_t)bh * 1000 + i];
  __syncthreads();
  for (int t = 0; t < 8; ++t) {
    float bv = -3.4e38f; int bi = 0x7fffffff;
    for (int i = tid; i < 1000; i += 128) {
      float v = vals[i];
      if (v > bv || (v == bv && i < bi)) { bv = v; bi = i; }
    }
    rv[tid] = bv; ri[tid] = bi;
    __syncthreads();
    for (int off = 64; off > 0; off >>= 1) {
      if (tid < off) {
        if (rv[tid + off] > rv[tid] ||
            (rv[tid + off] == rv[tid] && ri[tid + off] < ri[tid])) {
          rv[tid] = rv[tid + off]; ri[tid] = ri[tid + off];
        }
      }
      __syncthreads();
    }
    if (tid == 0) { idxs[bh * 8 + t] = ri[0]; vals[ri[0]] = -3.4e38f; }
    __syncthreads();
  }
}

// tok[bf, t, h*64+e] = k[idx[bf,h,t], h*64+e]
__global__ void gather_tok_kernel(const float* __restrict__ kbuf,
                                  const int* __restrict__ idxs,
                                  float* __restrict__ tok)
{
  const int row = blockIdx.x;  // 0..895 = bf*8 + t
  const int bf = row >> 3, t = row & 7;
  for (int d = threadIdx.x; d < 768; d += blockDim.x) {
    int h  = d >> 6;
    int id = idxs[(bf * 12 + h) * 8 + t];
    tok[(size_t)row * 768 + d] = kbuf[(size_t)id * 768 + d];
  }
}

// d_out rows: [bf*71 + 0..7] = attention out, [bf*71 + 8..70] = x_token
__global__ void scatter_xtok_kernel(const float* __restrict__ xt, float* __restrict__ out) {
  const int r = blockIdx.x;  // 0..7055 = bf*63 + p
  const int bf = r / 63, p = r % 63;
  float* dst = out + (size_t)(bf * 71 + 8 + p) * 768;
  const float* src = xt + (size_t)r * 768;
  for (int d = threadIdx.x; d < 768; d += blockDim.x) dst[d] = src[d];
}

__global__ void scatter_attn_kernel(const float* __restrict__ oa, float* __restrict__ out) {
  const int r = blockIdx.x;  // 0..895 = bf*8 + t
  const int bf = r >> 3, t = r & 7;
  float* dst = out + (size_t)(bf * 71 + t) * 768;
  const float* src = oa + (size_t)r * 768;
  for (int d = threadIdx.x; d < 768; d += blockDim.x) dst[d] = src[d];
}

// ---------------------------------------------------------------------------
extern "C" void kernel_launch(void* const* d_in, const int* in_sizes, int n_in,
                              void* d_out, int out_size, void* d_ws, size_t ws_size,
                              hipStream_t stream)
{
  (void)in_sizes; (void)n_in; (void)out_size; (void)ws_size;
  const float* x_inp  = (const float*)d_in[0];
  const float* mask   = (const float*)d_in[1];
  const float* W_feat = (const float*)d_in[2];
  const float* b_feat = (const float*)d_in[3];
  const float* W_bin  = (const float*)d_in[4];
  const float* b_bin  = (const float*)d_in[5];
  const float* W_g1   = (const float*)d_in[6];
  const float* b_g1   = (const float*)d_in[7];
  const float* W_g2   = (const float*)d_in[8];
  const float* b_g2   = (const float*)d_in[9];
  const float* W_fp   = (const float*)d_in[10];
  const float* b_fp   = (const float*)d_in[11];
  const float* wemb   = (const float*)d_in[12];  // (50257, 768)
  const float* W_map  = (const float*)d_in[13];  // (1000, 50257)
  const float* b_map  = (const float*)d_in[14];
  const float* W_q    = (const float*)d_in[15];
  const float* b_q    = (const float*)d_in[16];
  const float* W_k    = (const float*)d_in[17];
  const float* b_k    = (const float*)d_in[18];
  const float* W_o    = (const float*)d_in[19];
  const float* b_o    = (const float*)d_in[20];
  // d_in[21..23] = data_id, seq_len(512), stride(8): compile-time constants here.
  float* out = (float*)d_out;
  float* ws  = (float*)d_ws;

  constexpr size_t SZ_TOK = (size_t)7056 * 768;    // 5,419,008
  float* xe   = ws;                  // x_emb, later reused for x_token
  float* me   = ws + SZ_TOK;         // m_emb, later reused for q
  float* gg   = ws + 2 * SZ_TOK;     // gate logits -> h (in place)
  float* srcb = ws + 3 * SZ_TOK;     // src (1000, 768)
  float* kb   = srcb + 768000;       // k   (1000, 768)
  float* tmps = kb + 768000;         // (112,12,1000)
  int*   idxs = (int*)(tmps + 1344000);            // (112,12,8)
  float* tok  = (float*)(idxs + 10752);            // (896, 768)
  float* oat  = tok + 688128;                      // (896, 768)
  float* xn   = oat + 688128;                      // (112, 512)

  // 1. normalize + transpose
  norm_kernel<<<112, 256, 0, stream>>>(x_inp, mask, xn);
  // 2. patch embeddings (K=16)
  patch_embed_kernel<<<dim3(63, 112), 256, 0, stream>>>(xn, mask, W_feat, b_feat,
                                                        W_bin, b_bin, xe, me);
  // 3. gate logits: g = xe@Wg1^T + b_g1 + me@Wg2^T + b_g2
  gemm_bf16_kernel<1, true, false><<<dim3(6, 56), 256, 0, stream>>>(
      xe, W_g1, b_g1, gg, 7056, 768, 768, 768, 768, 768);
  gemm_bf16_kernel<1, true, true><<<dim3(6, 56), 256, 0, stream>>>(
      me, W_g2, b_g2, gg, 7056, 768, 768, 768, 768, 768);
  // 4. gated mix (in place over gg)
  gate_kernel<<<2048, 256, 0, stream>>>(gg, xe, me, SZ_TOK);
  // 5. x_token = h @ Wfp^T + b_fp   (into xe)
  gemm_bf16_kernel<1, true, false><<<dim3(6, 56), 256, 0, stream>>>(
      gg, W_fp, b_fp, xe, 7056, 768, 768, 768, 768, 768);
  scatter_xtok_kernel<<<7056, 256, 0, stream>>>(xe, out);
  // 6. q = x_token @ Wq^T + b_q   (into me)
  gemm_bf16_kernel<1, true, false><<<dim3(6, 56), 256, 0, stream>>>(
      xe, W_q, b_q, me, 7056, 768, 768, 768, 768, 768);
  // 7. src = W_map @ word_embeddings + b_map[:,None]   (per-row bias, K=50257)
  gemm_bf16_kernel<2, false, false><<<dim3(6, 8), 256, 0, stream>>>(
      W_map, wemb, b_map, srcb, 1000, 768, 50257, 50257, 768, 768);
  // 8. k = src @ Wk^T + b_k
  gemm_bf16_kernel<1, true, false><<<dim3(6, 8), 256, 0, stream>>>(
      srcb, W_k, b_k, kb, 1000, 768, 768, 768, 768, 768);
  // 9. attention score stats -> tmps
  attn_scores_kernel<<<dim3(112, 12), 128, 0, stream>>>(me, kb, tmps);
  // 10. top-8 tokens per (bf, h)
  topk_kernel<<<1344, 128, 0, stream>>>(tmps, idxs);
  // 11. gather tok
  gather_tok_kernel<<<896, 256, 0, stream>>>(kb, idxs, tok);
  // 12. out = tok @ Wo^T + b_o
  gemm_bf16_kernel<1, true, false><<<dim3(6, 7), 256, 0, stream>>>(
      tok, W_o, b_o, oat, 896, 768, 768, 768, 768, 768);
  scatter_attn_kernel<<<896, 256, 0, stream>>>(oat, out);
}
